// CopyGenerator_30889404793240
// MI455X (gfx1250) — matrix operand order repaired and checked by reading, not compiled, not run on hardware
//
#include <hip/hip_runtime.h>
#include <hip/hip_bf16.h>
#include <math.h>

#define TLEN   64
#define BSZ    32
#define SLEN   64
#define HSIZE  1024
#define VOCAB  32000
#define MROWS  (TLEN * BSZ)   // 2048
#define PAD_TOK 0
#define C_CONST 0.1712209f

#define NT 128   // vocab cols per block (8 waves * 16)
#define MTA 64   // rows per block, bf16-weight path (4 stacked tiles / wave)
#define MTB 32   // rows per block, fallback path
#define KT 32    // K per WMMA step
#define KB 64    // K per TDM stage (2 WMMA steps per buffer)

#define USE_TDM 1

typedef __attribute__((ext_vector_type(16))) __bf16   v16bf;
typedef __attribute__((ext_vector_type(8)))  float    v8f;
typedef __attribute__((ext_vector_type(4)))  unsigned v4u;
typedef __attribute__((ext_vector_type(8)))  unsigned v8u;

union Frag16 { v16bf v; unsigned u[8]; };

// f32 -> bf16 round-to-nearest-even, packed pair (used once, in pre-convert)
__device__ __forceinline__ unsigned pack_bf16_rne(float a, float b) {
  unsigned ua = __float_as_uint(a), ub = __float_as_uint(b);
  ua = (ua + 0x7FFFu + ((ua >> 16) & 1u)) >> 16;
  ub = (ub + 0x7FFFu + ((ub >> 16) & 1u)) >> 16;
  return (ua & 0xFFFFu) | (ub << 16);
}
// cheap truncating pack (fallback hot loop only)
__device__ __forceinline__ unsigned pack_bf16_trunc(float a, float b) {
  return (__float_as_uint(a) >> 16) | (__float_as_uint(b) & 0xFFFF0000u);
}

// ---------------------------------------------------------------------------
// Kernel 0: bulk f32 -> packed bf16 conversion (one-time, bandwidth bound)
// ---------------------------------------------------------------------------
__global__ __launch_bounds__(256)
void cvt_f32_bf16_kernel(const float4* __restrict__ in, uint2* __restrict__ out,
                         int n4) {
  const int i = blockIdx.x * 256 + threadIdx.x;
  if (i < n4) {
    const float4 f = in[i];
    out[i] = make_uint2(pack_bf16_rne(f.x, f.y), pack_bf16_rne(f.z, f.w));
  }
}

// ---------------------------------------------------------------------------
// Kernel 1: copy gate  cp[r] = sigmoid(hidden[r,:] . w_copy + b_copy)
// ---------------------------------------------------------------------------
__global__ __launch_bounds__(256)
void copy_gate_kernel(const float* __restrict__ hidden,
                      const float* __restrict__ w_copy,
                      const float* __restrict__ b_copy,
                      float* __restrict__ cp) {
  const int wave = threadIdx.x >> 5;
  const int lane = threadIdx.x & 31;
  const int row  = blockIdx.x * 8 + wave;
  const float* h = hidden + (size_t)row * HSIZE;
  float acc = 0.f;
  #pragma unroll 4
  for (int k = lane; k < HSIZE; k += 32) acc += h[k] * w_copy[k];
  #pragma unroll
  for (int off = 16; off > 0; off >>= 1) acc += __shfl_xor(acc, off, 32);
  if (lane == 0) cp[row] = 1.f / (1.f + __expf(-(acc + b_copy[0])));
}

// ---------------------------------------------------------------------------
// Kernel 2: copy attention -> p_c[t,b,s]
// ---------------------------------------------------------------------------
__global__ __launch_bounds__(256)
void attn_pc_kernel(const float* __restrict__ hidden,
                    const float* __restrict__ context,
                    const int*   __restrict__ src,
                    const float* __restrict__ cp,
                    float* __restrict__ p_c) {
  const int row = blockIdx.x;          // t*BSZ + b
  const int b   = row & (BSZ - 1);
  const int tid = threadIdx.x;
  const int s    = tid >> 2;
  const int part = tid & 3;

  const float* h = hidden  + (size_t)row * HSIZE;
  const float* c = context + (size_t)(s * BSZ + b) * HSIZE;
  float acc = 0.f;
  const int k0 = part * (HSIZE / 4);
  #pragma unroll 4
  for (int k = k0; k < k0 + HSIZE / 4; ++k) acc += h[k] * c[k];

  __shared__ float partial[256];
  __shared__ float sc[SLEN];
  partial[tid] = acc;
  __syncthreads();
  if (tid < SLEN) {
    float v = partial[tid*4] + partial[tid*4+1] + partial[tid*4+2] + partial[tid*4+3];
    if (src[b * SLEN + tid] == PAD_TOK) v = -INFINITY;
    sc[tid] = v;
  }
  __syncthreads();
  if (tid < SLEN) {
    float mx = -INFINITY;
    for (int i = 0; i < SLEN; ++i) mx = fmaxf(mx, sc[i]);
    float sum = 0.f;
    for (int i = 0; i < SLEN; ++i) sum += __expf(sc[i] - mx);
    p_c[row * SLEN + tid] = __expf(sc[tid] - mx) / sum * cp[row];
  }
}

#if USE_TDM
// Issue a TDM load of a 64(K) x 64(rows) bf16 A tile into LDS.
// g1 carries the constant tensor/tile geometry; g0 the per-call addresses.
__device__ __forceinline__ void tdm_load_a(const unsigned* Hbf, int row0, int kb,
                                           unsigned lds_addr, const v8u& g1) {
  const unsigned long long ga =
      (unsigned long long)(uintptr_t)Hbf + ((size_t)row0 * (HSIZE / 2) + (kb >> 1)) * 4ull;
  v4u g0;
  g0.x = 1u;                                              // count=1, user mode
  g0.y = lds_addr;                                        // lds_addr (bytes)
  g0.z = (unsigned)(ga & 0xFFFFFFFFu);                    // global_addr[31:0]
  g0.w = (unsigned)((ga >> 32) & 0x01FFFFFFu) | 0x80000000u; // addr[56:32] | type=2
  asm volatile("tensor_load_to_lds %0, %1" :: "s"(g0), "s"(g1) : "memory");
}
#endif

// ---------------------------------------------------------------------------
// Kernel 3A: logits GEMM with pre-converted bf16 A/B.
// Block: 256 thr (8 waves), tile 64(M) x 128(N), K stepped 64 per TDM stage.
// Double-buffered LDS A tiles: the Tensor Data Mover fills the next buffer
// while all 8 waves run 8 WMMAs against the current one (TENSORcnt pipelining).
// ---------------------------------------------------------------------------
__global__ __launch_bounds__(256)
void gemm_bf16_kernel(const unsigned* __restrict__ Hbf,   // MROWS x HSIZE/2 dwords
                      const unsigned* __restrict__ Wbf,   // VOCAB x HSIZE/2 dwords
                      float* __restrict__ out) {
  __shared__ unsigned As[2][MTA * (KB / 2)];   // two 8 KB buffers: [row][kdword]
  const int tid  = threadIdx.x;
  const int wave = tid >> 5;
  const int lane = tid & 31;
  const int row0 = blockIdx.y * MTA;
  const int kh   = lane >> 4;          // K half within a 32-step
  const int m    = lane & 15;
  const int ncol = blockIdx.x * NT + wave * 16 + m;

  v8f acc0 = {}, acc1 = {}, acc2 = {}, acc3 = {};
  const unsigned* wrow = Wbf + (size_t)ncol * (HSIZE / 2);

#if USE_TDM
  const unsigned lds0 = (unsigned)(uintptr_t)(&As[0][0]);
  const unsigned lds1 = (unsigned)(uintptr_t)(&As[1][0]);
  // D# group 1 (constant): data_size=2B; tensor_dim0=HSIZE @79:48,
  // tensor_dim1=MROWS @111:80, tile_dim0=KB @127:112, tile_dim1=MTA @143:128,
  // tensor_dim0_stride=HSIZE @207:160.
  const v8u g1 = { 0x00010000u,
                   (unsigned)(HSIZE & 0xFFFF) << 16,
                   ((unsigned)(HSIZE >> 16) & 0xFFFFu) | ((unsigned)(MROWS & 0xFFFF) << 16),
                   ((unsigned)(MROWS >> 16) & 0xFFFFu) | ((unsigned)KB << 16),
                   (unsigned)MTA,
                   (unsigned)HSIZE,
                   0u, 0u };
  if (wave == 0) tdm_load_a(Hbf, row0, 0, lds0, g1);      // prologue fill buf0
#endif

  for (int kb = 0; kb < HSIZE; kb += KB) {
    const int cur = (kb >> 6) & 1;
#if USE_TDM
    if (wave == 0) __builtin_amdgcn_s_wait_tensorcnt(0);  // buf[cur] landed
    __syncthreads();                                      // visible to all; buf[1-cur] free
    if (wave == 0 && (kb + KB) < HSIZE)
      tdm_load_a(Hbf, row0, kb + KB, cur ? lds0 : lds1, g1); // overlap with compute
#else
    __syncthreads();
    {
      // cooperative stage: 64 rows x 32 dwords, two uint4 per thread
      const int r = tid >> 2, q = tid & 3;
      const unsigned* hsrc = Hbf + (size_t)(row0 + r) * (HSIZE / 2) + (kb >> 1) + q * 8;
      *(uint4*)(&As[cur][r * 32 + q * 8])     = *(const uint4*)(hsrc);
      *(uint4*)(&As[cur][r * 32 + q * 8 + 4]) = *(const uint4*)(hsrc + 4);
    }
    __syncthreads();
#endif

    #pragma unroll
    for (int t = 0; t < 2; ++t) {                         // two 32-K WMMA steps
      const int k0 = kb + t * KT;
      // B fragment: lanes 0-15 K=k0..7,k0+16..23; lanes 16-31 K=k0+8..15,k0+24..31 (8 dwords)
      Frag16 bf;
      {
        const uint4* wp = (const uint4*)(wrow + (k0 >> 1) + kh * 4);
        if (k0 + KT < HSIZE) __builtin_prefetch(wp + 4, 0, 1);
        const uint4 w0 = wp[0], w1 = wp[2];
        bf.u[0] = w0.x; bf.u[1] = w0.y; bf.u[2] = w0.z; bf.u[3] = w0.w;
        bf.u[4] = w1.x; bf.u[5] = w1.y; bf.u[6] = w1.z; bf.u[7] = w1.w;
      }
      // A fragments (16-bit A layout) for the 4 stacked row tiles
      Frag16 a0, a1, a2, a3;
      const int kd = t * 16 + kh * 4;                     // dword offset in row
      #pragma unroll
      for (int j = 0; j < 4; ++j) {
        a0.u[j]     = As[cur][(m     ) * 32 + kd + j];
        a0.u[4 + j] = As[cur][(m     ) * 32 + 8 + kd + j];
        a1.u[j]     = As[cur][(m + 16) * 32 + kd + j];
        a1.u[4 + j] = As[cur][(m + 16) * 32 + 8 + kd + j];
        a2.u[j]     = As[cur][(m + 32) * 32 + kd + j];
        a2.u[4 + j] = As[cur][(m + 32) * 32 + 8 + kd + j];
        a3.u[j]     = As[cur][(m + 48) * 32 + kd + j];
        a3.u[4 + j] = As[cur][(m + 48) * 32 + 8 + kd + j];
      }
      acc0 = __builtin_amdgcn_wmma_f32_16x16x32_bf16(false, a0.v, false, bf.v, (short)0, acc0, false, false);
      acc1 = __builtin_amdgcn_wmma_f32_16x16x32_bf16(false, a1.v, false, bf.v, (short)0, acc1, false, false);
      acc2 = __builtin_amdgcn_wmma_f32_16x16x32_bf16(false, a2.v, false, bf.v, (short)0, acc2, false, false);
      acc3 = __builtin_amdgcn_wmma_f32_16x16x32_bf16(false, a3.v, false, bf.v, (short)0, acc3, false, false);
    }
  }

  // C/D layout: lanes 0-15 -> M=j, lanes 16-31 -> M=8+j; N = lane&15
  const int rbase = row0 + kh * 8;
  #pragma unroll
  for (int j = 0; j < 8; ++j) {
    out[(size_t)(rbase + j)      * VOCAB + ncol] = acc0[j];
    out[(size_t)(rbase + 16 + j) * VOCAB + ncol] = acc1[j];
    out[(size_t)(rbase + 32 + j) * VOCAB + ncol] = acc2[j];
    out[(size_t)(rbase + 48 + j) * VOCAB + ncol] = acc3[j];
  }
}

// ---------------------------------------------------------------------------
// Kernel 3B: fallback GEMM (on-the-fly f32->bf16 truncation), ws too small
// ---------------------------------------------------------------------------
__global__ __launch_bounds__(256)
void gemm_f32src_kernel(const float* __restrict__ hidden,
                        const float* __restrict__ W_gen,
                        float* __restrict__ out) {
  __shared__ unsigned As[MTB * 16];
  const int tid  = threadIdx.x;
  const int wave = tid >> 5;
  const int lane = tid & 31;
  const int row0 = blockIdx.y * MTB;
  const int kh   = lane >> 4;
  const int m    = lane & 15;
  const int ncol = blockIdx.x * NT + wave * 16 + m;

  v8f acc0 = {}, acc1 = {};
  const float4* wrow = (const float4*)(W_gen + (size_t)ncol * HSIZE);

  for (int k0 = 0; k0 < HSIZE; k0 += KT) {
    __syncthreads();
    {
      int p = tid * 2;
      #pragma unroll
      for (int q = 0; q < 2; ++q, ++p) {
        const int r = p >> 4, c = p & 15;
        const float2 f = *(const float2*)(hidden + (size_t)(row0 + r) * HSIZE + k0 + c * 2);
        As[r * 16 + c] = pack_bf16_trunc(f.x, f.y);
      }
    }
    __syncthreads();

    Frag16 bf;
    {
      const float4* wp = wrow + ((k0 + kh * 8) >> 2);
      if (k0 + KT < HSIZE) __builtin_prefetch(wp + 8, 0, 1);
      #pragma unroll
      for (int j = 0; j < 4; ++j) {
        const float4 f = wp[j + (j >> 1) * 2];
        bf.u[2*j]   = pack_bf16_trunc(f.x, f.y);
        bf.u[2*j+1] = pack_bf16_trunc(f.z, f.w);
      }
    }
    Frag16 a0, a1;
    #pragma unroll
    for (int j = 0; j < 4; ++j) {
      a0.u[j]     = As[m * 16 + kh * 4 + j];
      a0.u[4 + j] = As[m * 16 + 8 + kh * 4 + j];
      a1.u[j]     = As[(m + 16) * 16 + kh * 4 + j];
      a1.u[4 + j] = As[(m + 16) * 16 + 8 + kh * 4 + j];
    }
    acc0 = __builtin_amdgcn_wmma_f32_16x16x32_bf16(false, a0.v, false, bf.v, (short)0, acc0, false, false);
    acc1 = __builtin_amdgcn_wmma_f32_16x16x32_bf16(false, a1.v, false, bf.v, (short)0, acc1, false, false);
  }
  const int rbase = row0 + kh * 8;
  #pragma unroll
  for (int j = 0; j < 8; ++j) {
    out[(size_t)(rbase + j)      * VOCAB + ncol] = acc0[j];
    out[(size_t)(rbase + 16 + j) * VOCAB + ncol] = acc1[j];
  }
}

// ---------------------------------------------------------------------------
// Kernel 4: fused softmax + copy-scatter + log. One block per row; dense
// 128 KB LDS vocab accumulator (dynamic LDS; 320 KB/WG on CDNA5).
// ---------------------------------------------------------------------------
__global__ __launch_bounds__(256)
void softmax_scatter_log_kernel(float* __restrict__ out,
                                const float* __restrict__ b_gen,
                                const int*   __restrict__ src,
                                const float* __restrict__ cp,
                                const float* __restrict__ p_c) {
  extern __shared__ float smem[];
  float* padd = smem;
  float* redm = smem + VOCAB;
  float* reds = redm + 256;
  const int row = blockIdx.x;
  const int b   = row & (BSZ - 1);
  const int tid = threadIdx.x;
  float* orow = out + (size_t)row * VOCAB;

  for (int v = tid; v < VOCAB; v += 256) padd[v] = 0.f;
  __syncthreads();
  if (tid < SLEN) {
    const int idx = src[b * SLEN + tid];
    atomicAdd(&padd[idx], p_c[row * SLEN + tid]);   // ds_add_f32
  }
  __syncthreads();

  float m = -INFINITY, ssum = 0.f;
  for (int v = tid; v < VOCAB; v += 256) {
    const float x = orow[v] + b_gen[v];
    if (x > m) { ssum = ssum * __expf(m - x) + 1.f; m = x; }
    else       { ssum += __expf(x - m); }
  }
  redm[tid] = m; reds[tid] = ssum;
  __syncthreads();
  #pragma unroll
  for (int off = 128; off > 0; off >>= 1) {
    if (tid < off) {
      const float m1 = redm[tid], s1 = reds[tid];
      const float m2 = redm[tid + off], s2 = reds[tid + off];
      const float M = fmaxf(m1, m2);
      reds[tid] = s1 * __expf(m1 - M) + s2 * __expf(m2 - M);
      redm[tid] = M;
    }
    __syncthreads();
  }
  const float M = redm[0];
  const float scale = (1.f - cp[row]) / reds[0];
  for (int v = tid; v < VOCAB; v += 256) {
    const float x = orow[v] + b_gen[v];
    const float p = __expf(x - M) * scale + padd[v];
    orow[v] = __logf(p) + C_CONST;
  }
}

// ---------------------------------------------------------------------------
extern "C" void kernel_launch(void* const* d_in, const int* in_sizes, int n_in,
                              void* d_out, int out_size, void* d_ws, size_t ws_size,
                              hipStream_t stream) {
  (void)in_sizes; (void)n_in; (void)out_size;
  const float* hidden  = (const float*)d_in[0];
  const float* context = (const float*)d_in[1];
  const int*   src     = (const int*)  d_in[2];
  const float* W_gen   = (const float*)d_in[3];
  const float* b_gen   = (const float*)d_in[4];
  const float* w_copy  = (const float*)d_in[5];
  const float* b_copy  = (const float*)d_in[6];
  float* out = (float*)d_out;

  char* ws = (char*)d_ws;
  const size_t off_cp  = 0;
  const size_t off_pc  = off_cp + (size_t)MROWS * sizeof(float);
  const size_t off_hbf = off_pc + (size_t)MROWS * SLEN * sizeof(float);
  const size_t off_wbf = off_hbf + (size_t)MROWS * HSIZE * 2;   // bf16
  const size_t need    = off_wbf + (size_t)VOCAB * HSIZE * 2;   // ~70 MB

  float* cp  = (float*)(ws + off_cp);
  float* p_c = (float*)(ws + off_pc);

  copy_gate_kernel<<<MROWS / 8, 256, 0, stream>>>(hidden, w_copy, b_copy, cp);
  attn_pc_kernel<<<MROWS, 256, 0, stream>>>(hidden, context, src, cp, p_c);

  if (ws_size >= need) {
    unsigned* Hbf = (unsigned*)(ws + off_hbf);
    unsigned* Wbf = (unsigned*)(ws + off_wbf);
    const int h4 = MROWS * HSIZE / 4;
    const int w4 = VOCAB * HSIZE / 4;
    cvt_f32_bf16_kernel<<<(h4 + 255) / 256, 256, 0, stream>>>((const float4*)hidden, (uint2*)Hbf, h4);
    cvt_f32_bf16_kernel<<<(w4 + 255) / 256, 256, 0, stream>>>((const float4*)W_gen,  (uint2*)Wbf, w4);
    gemm_bf16_kernel<<<dim3(VOCAB / NT, MROWS / MTA), 256, 0, stream>>>(Hbf, Wbf, out);
  } else {
    gemm_f32src_kernel<<<dim3(VOCAB / NT, MROWS / MTB), 256, 0, stream>>>(hidden, W_gen, out);
  }

  const size_t smem_bytes = (VOCAB + 512) * sizeof(float);
  softmax_scatter_log_kernel<<<MROWS, 256, smem_bytes, stream>>>(out, b_gen, src, cp, p_c);
}
